// AttentionBlock_83640193122859
// MI455X (gfx1250) — matrix operand order repaired and checked by reading, not compiled, not run on hardware
//
#include <hip/hip_runtime.h>

typedef __bf16 bf16;
typedef unsigned short u16;
typedef __attribute__((ext_vector_type(16))) __bf16 v16bf;
typedef __attribute__((ext_vector_type(8)))  __bf16 v8bf;
typedef __attribute__((ext_vector_type(8)))  float  v8f;
typedef int v4i_vs __attribute__((vector_size(16)));   // matches builtin's pointee type

#define WMMA_BF16(a, b, c) \
  __builtin_amdgcn_wmma_f32_16x16x32_bf16(false, (a), false, (b), (short)0, (c), false, false)

// ---- CDNA5 async global->LDS copy (ASYNCcnt path), with safe fallback ------
#if defined(__AMDGCN__) && __has_builtin(__builtin_amdgcn_global_load_async_to_lds_b128)
#define HAVE_ASYNC_LDS 1
#else
#define HAVE_ASYNC_LDS 0
#endif

#if HAVE_ASYNC_LDS
__device__ __forceinline__ void async_copy16(const void* g, void* l) {
  // (global src v4i*, LDS dst v4i*, imm offset, imm cpol)
  __builtin_amdgcn_global_load_async_to_lds_b128(
      (__attribute__((address_space(1))) v4i_vs*)(unsigned long long)g,
      (__attribute__((address_space(3))) v4i_vs*)l, 0, 0);
}
#endif

__device__ __forceinline__ void wait_async0() {
#if HAVE_ASYNC_LDS
#if __has_builtin(__builtin_amdgcn_s_wait_asynccnt)
  __builtin_amdgcn_s_wait_asynccnt(0);
#else
  asm volatile("s_wait_asynccnt 0" ::: "memory");
#endif
#endif
}

__device__ __forceinline__ u16 f2bf(float f) {
  unsigned int u = __float_as_uint(f);
  u += 0x7FFFu + ((u >> 16) & 1u);           // round-to-nearest-even
  return (u16)(u >> 16);
}

// Assemble a v16bf fragment from two contiguous 16-byte chunks.
__device__ __forceinline__ v16bf frag16(const bf16* p0, const bf16* p1) {
  v8bf a = *(const v8bf*)p0;
  v8bf b = *(const v8bf*)p1;
  v16bf r;
#pragma unroll
  for (int i = 0; i < 8; ++i) { r[i] = a[i]; r[8 + i] = b[i]; }
  return r;
}

// ---------------------------------------------------------------------------
// fp32 -> bf16 conversion (weights)
// ---------------------------------------------------------------------------
__global__ __launch_bounds__(256) void cvt_f32_bf16(const float* __restrict__ in,
                                                    u16* __restrict__ out, int n) {
  int i = blockIdx.x * 256 + threadIdx.x;
  if (i < n) out[i] = f2bf(in[i]);
}

// ---------------------------------------------------------------------------
// LayerNorm over C=512, one row per 256-thread block, bf16 output
// ---------------------------------------------------------------------------
__global__ __launch_bounds__(256) void layernorm_to_bf16(
    const float* __restrict__ x, const float* __restrict__ g,
    const float* __restrict__ be, u16* __restrict__ out) {
  __shared__ float red[16];
  int row = blockIdx.x;
  int tid = threadIdx.x;
  const float* xr = x + (size_t)row * 512;
  float v0 = xr[tid], v1 = xr[tid + 256];

  float s = v0 + v1;
#pragma unroll
  for (int off = 16; off; off >>= 1) s += __shfl_xor(s, off, 32);
  int lane = tid & 31, w = tid >> 5;
  if (!lane) red[w] = s;
  __syncthreads();
  float mean = 0.f;
#pragma unroll
  for (int i = 0; i < 8; ++i) mean += red[i];
  mean *= (1.0f / 512.0f);

  float d0 = v0 - mean, d1 = v1 - mean;
  float sq = d0 * d0 + d1 * d1;
#pragma unroll
  for (int off = 16; off; off >>= 1) sq += __shfl_xor(sq, off, 32);
  if (!lane) red[8 + w] = sq;   // disjoint from red[0..7] reads above
  __syncthreads();
  float var = 0.f;
#pragma unroll
  for (int i = 0; i < 8; ++i) var += red[8 + i];
  var *= (1.0f / 512.0f);
  float inv = rsqrtf(var + 1e-5f);

  out[(size_t)row * 512 + tid]       = f2bf(d0 * inv * g[tid] + be[tid]);
  out[(size_t)row * 512 + tid + 256] = f2bf(d1 * inv * g[tid + 256] + be[tid + 256]);
}

// ---------------------------------------------------------------------------
// bf16 WMMA GEMM: 128x64 block tile, 128 threads (4 waves), each wave a 32x64
// strip (2 row-groups x 4 col-tiles = 8 WMMAs per 32-wide k-step).
//   y = (HAS_RES? resid : 0) + alpha * (A@B + (HAS_BIAS? bias[col] : 0))
// A tile staged via async global->LDS (no transpose needed); B tile staged
// transposed through VGPRs. Strides in elements; batched via blockIdx.z.
// ---------------------------------------------------------------------------
template <bool HAS_BIAS, bool HAS_RES, bool RELU, bool OUT_BF16>
__global__ __launch_bounds__(128) void gemm_bf16_tile128(
    const u16* __restrict__ Ag, const u16* __restrict__ Bg,
    const float* __restrict__ bias, const float* __restrict__ resid,
    void* __restrict__ outp,
    int K, int lda, int ldb, int ldn,
    long long strideA, long long strideB, long long strideOut,
    int innerB, float alpha) {
  __shared__ bf16 As[128][40];  // [m][k], padded (rows 16B-aligned)
  __shared__ bf16 Bs[64][40];   // [n][k], transposed, padded

  int z  = blockIdx.z;
  int zA = z / innerB, zB = z % innerB;
  const bf16* A = (const bf16*)Ag + (long long)zA * strideA;
  const bf16* B = (const bf16*)Bg + (long long)zB * strideB;
  long long outOff = (long long)z * strideOut;

  int m0 = blockIdx.x * 128;
  int n0 = blockIdx.y * 64;
  int tid = threadIdx.x;
  int lane = tid & 31, wave = tid >> 5;
  int mq  = lane & 15;
  int kb  = (lane < 16) ? 0 : 16;   // A fragment K sub-offset
  int kb2 = (lane < 16) ? 0 : 16;   // B fragment K sub-offset

  v8f acc[2][4] = {};

  for (int k0 = 0; k0 < K; k0 += 32) {
    // ---- Stage A tile (128x32): thread t owns row t --------------------
    {
      const bf16* src = A + (size_t)(m0 + tid) * lda + k0;
      bf16* dst = &As[tid][0];
#if HAVE_ASYNC_LDS
      async_copy16(src,      dst);
      async_copy16(src + 8,  dst + 8);
      async_copy16(src + 16, dst + 16);
      async_copy16(src + 24, dst + 24);
#else
      v8bf x0 = *(const v8bf*)src;
      v8bf x1 = *(const v8bf*)(src + 8);
      v8bf x2 = *(const v8bf*)(src + 16);
      v8bf x3 = *(const v8bf*)(src + 24);
      *(v8bf*)(dst)      = x0;
      *(v8bf*)(dst + 8)  = x1;
      *(v8bf*)(dst + 16) = x2;
      *(v8bf*)(dst + 24) = x3;
#endif
      if (k0 + 32 < K) __builtin_prefetch(src + 32, 0, 0);
    }
    // ---- Stage B tile (32x64) transposed into Bs[n][k] ------------------
    {
      int kr = tid >> 2, nc = (tid & 3) * 16;
      const bf16* src = B + (size_t)(k0 + kr) * ldb + n0 + nc;
      v8bf x0 = *(const v8bf*)src;
      v8bf x1 = *(const v8bf*)(src + 8);
#pragma unroll
      for (int i = 0; i < 8; ++i) {
        Bs[nc + i][kr]     = x0[i];
        Bs[nc + 8 + i][kr] = x1[i];
      }
      if (k0 + 32 < K) __builtin_prefetch(src + (size_t)32 * ldb, 0, 0);
    }
    wait_async0();
    __syncthreads();

    // ---- 8 WMMAs: 2 row-groups x 4 col-tiles ---------------------------
    const bf16* ar0 = &As[32 * wave + mq][0];
    const bf16* ar1 = &As[32 * wave + 16 + mq][0];
    v16bf af0 = frag16(ar0 + kb, ar0 + kb + 8);
    v16bf af1 = frag16(ar1 + kb, ar1 + kb + 8);
#pragma unroll
    for (int nt = 0; nt < 4; ++nt) {
      const bf16* brow = &Bs[16 * nt + mq][0];
      v16bf bfr = frag16(brow + kb2, brow + kb2 + 8);
      acc[0][nt] = WMMA_BF16(af0, bfr, acc[0][nt]);
      acc[1][nt] = WMMA_BF16(af1, bfr, acc[1][nt]);
    }
    __syncthreads();
  }

  // ---- Branch-free epilogue. C layout: col = lane&15, row = r + 8*(lane>=16)
  float* outF = (float*)outp;
  u16*   outB = (u16*)outp;
  int rowBase = m0 + 32 * wave + ((lane & 16) ? 8 : 0);
#pragma unroll
  for (int gsel = 0; gsel < 2; ++gsel) {
#pragma unroll
    for (int nt = 0; nt < 4; ++nt) {
      int col = n0 + 16 * nt + (lane & 15);
      float bv = HAS_BIAS ? bias[col] : 0.f;
#pragma unroll
      for (int r = 0; r < 8; ++r) {
        int row = rowBase + 16 * gsel + r;
        float y = (acc[gsel][nt][r] + bv) * alpha;
        long long idx = outOff + (long long)row * ldn + col;
        if (HAS_RES) y += resid[idx];
        if (RELU)    y = fmaxf(y, 0.f);
        if (OUT_BF16) outB[idx] = f2bf(y);
        else          outF[idx] = y;
      }
    }
  }
}

// ---------------------------------------------------------------------------
// Flash attention: one wave per (b, h, 16-query tile). Online softmax over
// 32-key blocks with causal masking. q/k/v bf16 [B,H,T,64]; out bf16 [B,T,512]
// with heads concatenated (col = h*64 + d). Scores scaled by C^-0.5.
// ---------------------------------------------------------------------------
__global__ __launch_bounds__(32) void flash_attn(
    const u16* __restrict__ qg, const u16* __restrict__ kg,
    const u16* __restrict__ vg, u16* __restrict__ attn, int T) {
  __shared__ bf16 Pl[16][40];   // probs tile [m][k], padded
  __shared__ bf16 Vt[64][40];   // V^T tile  [d][k], padded

  int qi = blockIdx.x, h = blockIdx.y, b = blockIdx.z;
  int H = gridDim.y;
  int lane = threadIdx.x;
  size_t hd = ((size_t)(b * H + h)) * (size_t)T * 64;
  const bf16* Q  = (const bf16*)qg + hd;
  const bf16* Kk = (const bf16*)kg + hd;
  const bf16* V  = (const bf16*)vg + hd;

  int mq  = lane & 15;
  int kb  = (lane < 16) ? 0 : 16;
  int kb2 = (lane < 16) ? 0 : 16;
  int tq  = qi * 16;

  // Q fragments (A-matrix 16x32, d split 0..31 / 32..63)
  const bf16* qrow = Q + (size_t)(tq + mq) * 64;
  v16bf aq0 = frag16(qrow + kb,      qrow + kb + 8);
  v16bf aq1 = frag16(qrow + 32 + kb, qrow + 40 + kb);

  float mrow[8], lrow[8];
#pragma unroll
  for (int r = 0; r < 8; ++r) { mrow[r] = -1e30f; lrow[r] = 0.f; }
  v8f o[4] = {};

  const float sm = 0.044194173824159216f;  // 512^-0.5
  int nkb = (tq + 15) / 32 + 1;

  for (int blk = 0; blk < nkb; ++blk) {
    int s0 = blk * 32;
    // S = Q @ K^T, two 16x16 tiles (keys s0..s0+15 and s0+16..s0+31)
    v8f c0 = {}, c1 = {};
    {
      const bf16* kr0 = Kk + (size_t)(s0 + mq) * 64;
      const bf16* kr1 = Kk + (size_t)(s0 + 16 + mq) * 64;
      v16bf bk;
      bk = frag16(kr0 + kb2,      kr0 + kb2 + 8);      c0 = WMMA_BF16(aq0, bk, c0);
      bk = frag16(kr0 + 32 + kb2, kr0 + 32 + kb2 + 8); c0 = WMMA_BF16(aq1, bk, c0);
      bk = frag16(kr1 + kb2,      kr1 + kb2 + 8);      c1 = WMMA_BF16(aq0, bk, c1);
      bk = frag16(kr1 + 32 + kb2, kr1 + 32 + kb2 + 8); c1 = WMMA_BF16(aq1, bk, c1);
    }
    // Stage V^T for this key block: lane t owns key row s0+t
    {
      const bf16* vrow = V + (size_t)(s0 + lane) * 64;
#pragma unroll
      for (int d = 0; d < 64; ++d) Vt[d][lane] = vrow[d];
    }
    __syncthreads();

    // Online softmax update (half-wave reductions: rows 0-7 in lanes 0-15,
    // rows 8-15 in lanes 16-31, one row per accumulator element)
    float alf[8];
    int colg0 = s0 + (lane & 15);
    int colg1 = colg0 + 16;
#pragma unroll
    for (int r = 0; r < 8; ++r) {
      int row = tq + r + ((lane & 16) ? 8 : 0);
      float s0v = c0[r] * sm; if (colg0 > row) s0v = -1e30f;
      float s1v = c1[r] * sm; if (colg1 > row) s1v = -1e30f;
      float mx = fmaxf(s0v, s1v);
#pragma unroll
      for (int off = 1; off < 16; off <<= 1) mx = fmaxf(mx, __shfl_xor(mx, off, 32));
      float mnew = fmaxf(mrow[r], mx);
      float al = __expf(mrow[r] - mnew);
      float p0 = __expf(s0v - mnew);
      float p1 = __expf(s1v - mnew);
      float ps = p0 + p1;
#pragma unroll
      for (int off = 1; off < 16; off <<= 1) ps += __shfl_xor(ps, off, 32);
      lrow[r] = lrow[r] * al + ps;
      mrow[r] = mnew;
      alf[r] = al;
      int prow = r + ((lane & 16) ? 8 : 0);
      Pl[prow][lane & 15]        = (bf16)p0;
      Pl[prow][16 + (lane & 15)] = (bf16)p1;
    }
    __syncthreads();

    // Rescale accumulators, then O += P @ V
#pragma unroll
    for (int dt = 0; dt < 4; ++dt)
#pragma unroll
      for (int r = 0; r < 8; ++r) o[dt][r] *= alf[r];

    const bf16* prow = &Pl[mq][0];
    v16bf pa = frag16(prow + kb, prow + kb + 8);
#pragma unroll
    for (int dt = 0; dt < 4; ++dt) {
      const bf16* vtrow = &Vt[16 * dt + mq][0];
      v16bf bv = frag16(vtrow + kb2, vtrow + kb2 + 8);
      o[dt] = WMMA_BF16(pa, bv, o[dt]);
    }
    __syncthreads();   // protect Pl/Vt before next block overwrites
  }

  // Epilogue: out[b, tq+row, h*64 + d] = O / L, bf16
  u16* outp = attn + ((size_t)b * T + tq) * 512 + h * 64;
#pragma unroll
  for (int dt = 0; dt < 4; ++dt)
#pragma unroll
    for (int r = 0; r < 8; ++r) {
      int row = r + ((lane & 16) ? 8 : 0);
      float y = o[dt][r] / lrow[r];
      outp[(size_t)row * 512 + 16 * dt + (lane & 15)] = f2bf(y);
    }
}

// ---------------------------------------------------------------------------
// Host orchestration
// ---------------------------------------------------------------------------
extern "C" void kernel_launch(void* const* d_in, const int* in_sizes, int n_in,
                              void* d_out, int out_size, void* d_ws, size_t ws_size,
                              hipStream_t stream) {
  (void)in_sizes; (void)n_in; (void)out_size; (void)ws_size;

  const int Bb = 4, T = 2048, C = 512, H = 8, D = 64, F = 2048;
  const float SCALE = 7.8964440777149550f;   // 6 * 3^0.25

  const float* x      = (const float*)d_in[0];
  const float* wq     = (const float*)d_in[1];
  const float* wk     = (const float*)d_in[2];
  const float* wv     = (const float*)d_in[3];
  const float* w_proj = (const float*)d_in[4];
  const float* b_proj = (const float*)d_in[5];
  const float* w1     = (const float*)d_in[6];
  const float* b1     = (const float*)d_in[7];
  const float* w2     = (const float*)d_in[8];
  const float* b2     = (const float*)d_in[9];
  const float* g1     = (const float*)d_in[10];
  const float* be1    = (const float*)d_in[11];
  const float* g2     = (const float*)d_in[12];
  const float* be2    = (const float*)d_in[13];

  char* ws = (char*)d_ws;
  // bf16 weights: 6 MB
  u16* wqb = (u16*)(ws + 0);          // 8*512*64
  u16* wkb = (u16*)(ws + 524288);
  u16* wvb = (u16*)(ws + 1048576);
  u16* wpb = (u16*)(ws + 1572864);    // 512*512
  u16* w1b = (u16*)(ws + 2097152);    // 512*2048
  u16* w2b = (u16*)(ws + 4194304);    // 2048*512
  // activations
  u16*   xnb   = (u16*)(ws + 6291456);            // [8192,512] bf16 (ln1, reused for ln2)
  float* x1f   = (float*)(ws + 14680064);         // [8192,512] f32 (post-attn residual)
  u16*   qb    = (u16*)(ws + 31457280);           // [B,H,T,64] bf16
  u16*   kvb   = (u16*)(ws + 39845888);
  u16*   vvb   = (u16*)(ws + 48234496);
  u16*   attnb = (u16*)(ws + 56623104);           // [B,T,512] bf16
  u16*   hb    = (u16*)(ws + 31457280);           // [8192,2048] bf16, overlays q/k/v/attn

  // 1) weight conversions
  cvt_f32_bf16<<<(H*C*D + 255) / 256, 256, 0, stream>>>(wq, wqb, H*C*D);
  cvt_f32_bf16<<<(H*C*D + 255) / 256, 256, 0, stream>>>(wk, wkb, H*C*D);
  cvt_f32_bf16<<<(H*C*D + 255) / 256, 256, 0, stream>>>(wv, wvb, H*C*D);
  cvt_f32_bf16<<<(C*C   + 255) / 256, 256, 0, stream>>>(w_proj, wpb, C*C);
  cvt_f32_bf16<<<(C*F   + 255) / 256, 256, 0, stream>>>(w1, w1b, C*F);
  cvt_f32_bf16<<<(F*C   + 255) / 256, 256, 0, stream>>>(w2, w2b, F*C);

  // 2) LN1
  layernorm_to_bf16<<<Bb * T, 256, 0, stream>>>(x, g1, be1, xnb);

  // 3) QKV projections, scaled by SCALE: per (b,h) GEMM M=T K=C N=D
  dim3 gq(T / 128, 1, Bb * H);
  gemm_bf16_tile128<false, false, false, true><<<gq, 128, 0, stream>>>(
      xnb, wqb, nullptr, nullptr, qb,
      C, C, D, D, (long long)T * C, (long long)C * D, (long long)T * D, H, SCALE);
  gemm_bf16_tile128<false, false, false, true><<<gq, 128, 0, stream>>>(
      xnb, wkb, nullptr, nullptr, kvb,
      C, C, D, D, (long long)T * C, (long long)C * D, (long long)T * D, H, SCALE);
  gemm_bf16_tile128<false, false, false, true><<<gq, 128, 0, stream>>>(
      xnb, wvb, nullptr, nullptr, vvb,
      C, C, D, D, (long long)T * C, (long long)C * D, (long long)T * D, H, SCALE);

  // 4) causal flash attention -> attnb [B,T,512] (heads concatenated)
  flash_attn<<<dim3(T / 16, H, Bb), 32, 0, stream>>>(qb, kvb, vvb, attnb, T);

  // 5) output projection + bias + residual(x) -> x1 (fp32)
  dim3 gp((Bb * T) / 128, C / 64, 1);
  gemm_bf16_tile128<true, true, false, false><<<gp, 128, 0, stream>>>(
      attnb, wpb, b_proj, x, x1f,
      C, C, C, C, 0, 0, 0, 1, 1.0f);

  // 6) LN2
  layernorm_to_bf16<<<Bb * T, 256, 0, stream>>>(x1f, g2, be2, xnb);

  // 7) FFN1: relu(xn @ w1 + b1) -> hb (bf16)
  dim3 gf1((Bb * T) / 128, F / 64, 1);
  gemm_bf16_tile128<true, false, true, true><<<gf1, 128, 0, stream>>>(
      xnb, w1b, b1, nullptr, hb,
      C, C, F, F, 0, 0, 0, 1, 1.0f);

  // 8) FFN2: x1 + SCALE*(h @ w2 + b2) -> d_out (fp32)
  dim3 gf2((Bb * T) / 128, C / 64, 1);
  gemm_bf16_tile128<true, true, false, false><<<gf2, 128, 0, stream>>>(
      hb, w2b, b2, x1f, d_out,
      F, F, C, C, 0, 0, 0, 1, SCALE);
}
